// ChamferSimilarity_66718021976303
// MI455X (gfx1250) — matrix operand order repaired and checked by reading, not compiled, not run on hardware
//
#include <hip/hip_runtime.h>

typedef __attribute__((ext_vector_type(16))) _Float16 v16h;
typedef __attribute__((ext_vector_type(8)))  _Float16 v8h;
typedef __attribute__((ext_vector_type(8)))  float    v8f;
typedef __attribute__((ext_vector_type(4)))  unsigned int tdm_g0_t;
typedef __attribute__((ext_vector_type(8)))  int          tdm_g1_t;
typedef __attribute__((ext_vector_type(4)))  int          tdm_g2_t;

#define CH_NEG_INF (-1.0e9f)
#define CH_VALID_THRESH (-1.0e8f)

// padded LDS row stride (halves): 128 data + 8 pad -> 272B rows, bank-conflict-free
#define ROW_STRIDE 136

// LDS partition (bytes)
#define LDS_DTILE_OFF 0
#define LDS_DTILE_SZ  (256 * ROW_STRIDE * 2)          // 69632
#define LDS_QTILE_OFF (LDS_DTILE_OFF + LDS_DTILE_SZ)  // 69632
#define LDS_QTILE_SZ  (32 * ROW_STRIDE * 2)           // 8704
#define LDS_SIMS_OFF  (LDS_QTILE_OFF + LDS_QTILE_SZ)  // 78336
#define LDS_SIMS_SZ   (32 * 257 * 4)                  // 32896
#define LDS_ROWPART_OFF (LDS_SIMS_OFF + LDS_SIMS_SZ)  // 111232
#define LDS_REDVAL_OFF  (LDS_ROWPART_OFF + 8 * 32 * 4)
#define LDS_REDCNT_OFF  (LDS_REDVAL_OFF + 256 * 4)
#define LDS_ROWVAL_OFF  (LDS_REDCNT_OFF + 256 * 4)
#define LDS_ROWCNT_OFF  (LDS_ROWVAL_OFF + 32 * 4)
#define LDS_TOTAL       (LDS_ROWCNT_OFF + 32 * 4)     // 114560

// ---------------------------------------------------------------------------
// prep: L2-normalize each 128-dim row, zero masked rows, convert to f16.
// ---------------------------------------------------------------------------
__global__ void __launch_bounds__(256)
chamfer_prep(const float* __restrict__ x, const int* __restrict__ mask,
             _Float16* __restrict__ y, int nrows) {
  const int wave = threadIdx.x >> 5;
  const int lane = threadIdx.x & 31;
  const int row  = blockIdx.x * 8 + wave;
  if (row >= nrows) return;
  const float* xr = x + (size_t)row * 128;
  float v0 = xr[lane], v1 = xr[lane + 32], v2 = xr[lane + 64], v3 = xr[lane + 96];
  float ss = v0 * v0 + v1 * v1 + v2 * v2 + v3 * v3;
#pragma unroll
  for (int m = 16; m >= 1; m >>= 1) ss += __shfl_xor(ss, m, 32);
  const float scale = (mask[row] != 0) ? (1.0f / sqrtf(fmaxf(ss, 1e-24f))) : 0.0f;
  _Float16* yr = y + (size_t)row * 128;
  yr[lane]      = (_Float16)(v0 * scale);
  yr[lane + 32] = (_Float16)(v1 * scale);
  yr[lane + 64] = (_Float16)(v2 * scale);
  yr[lane + 96] = (_Float16)(v3 * scale);
}

// ---------------------------------------------------------------------------
// TDM: 1-D DMA of n8 8-byte elements into LDS, inserting 16B of padding after
// every 256B (one 128-half row) so LDS rows land at a 272B stride.
// D# bit layout per CDNA5 ISA ch.8 (group0: count/lds_addr/global_addr/type,
// group1: data_size|pad fields, tensor_dim0, tile_dim0, stride).
// Toolchain here exposes the 6-arg builtin (clang-23 / therock-10.0 form).
// ---------------------------------------------------------------------------
#if __has_builtin(__builtin_amdgcn_tensor_load_to_lds)
#define CH_HAS_TDM 1
__device__ __forceinline__ void tdm_load_1d_padded(unsigned lds_off,
                                                   const void* gsrc,
                                                   unsigned n8) {
  const unsigned long long ga = (unsigned long long)(uintptr_t)gsrc;
  tdm_g0_t g0;
  g0[0] = 1u;                                    // count=1 (valid user D#)
  g0[1] = lds_off;                               // lds_addr
  g0[2] = (unsigned)(ga & 0xFFFFFFFFu);          // global_addr[31:0]
  g0[3] = (unsigned)((ga >> 32) & 0x01FFFFFFu)   // global_addr[56:32]
          | (2u << 30);                          // type=2 ("image")
  tdm_g1_t g1;
  g1[0] = (int)((3u << 16)                       // data_size = 8B
              | (1u << 20)                       // pad_enable
              | (5u << 22)                       // pad_interval: 64 DWORDs (256B)
              | (3u << 25));                     // pad_amount: 4 DWORDs (16B)
  g1[1] = (int)((n8 & 0xFFFFu) << 16);           // tensor_dim0[15:0]  (bits 63:48)
  g1[2] = (int)((n8 >> 16) & 0xFFFFu);           // tensor_dim0[31:16]; tensor_dim1=0
  g1[3] = (int)((n8 & 0xFFFFu) << 16);           // tile_dim0 (bits 127:112)
  g1[4] = 0;                                     // tile_dim1=0, tile_dim2=0
  g1[5] = (int)n8;                               // tensor_dim0_stride[31:0]
  g1[6] = 0;
  g1[7] = 0;
  tdm_g2_t z4 = {0, 0, 0, 0};
  tdm_g1_t z8 = {0, 0, 0, 0, 0, 0, 0, 0};
  __builtin_amdgcn_tensor_load_to_lds(g0, g1, z4, z4, z8, 0);
}
#else
#define CH_HAS_TDM 0
#endif

// ---------------------------------------------------------------------------
// pair kernel: one workgroup per (bq, bd). TDM stages q/d f16 tiles into
// padded LDS; 8 waves each own one 16-row M-tile and four 16-col N-tiles.
// ---------------------------------------------------------------------------
__global__ void __launch_bounds__(256)
chamfer_pair(const _Float16* __restrict__ qh, const _Float16* __restrict__ dh,
             const int* __restrict__ qmask, const int* __restrict__ dmask,
             float* __restrict__ out) {
  extern __shared__ char smem[];
  _Float16* dtile = (_Float16*)(smem + LDS_DTILE_OFF);
  _Float16* qtile = (_Float16*)(smem + LDS_QTILE_OFF);
  float* sims     = (float*)(smem + LDS_SIMS_OFF);
  float* rowpart  = (float*)(smem + LDS_ROWPART_OFF);
  float* redval   = (float*)(smem + LDS_REDVAL_OFF);
  float* redcnt   = (float*)(smem + LDS_REDCNT_OFF);
  float* rowval   = (float*)(smem + LDS_ROWVAL_OFF);
  float* rowcnt   = (float*)(smem + LDS_ROWCNT_OFF);

  const int bd = blockIdx.x;
  const int bq = blockIdx.y;
  const int tid  = threadIdx.x;
  const int wave = tid >> 5;
  const int lane = tid & 31;
  const int l16  = lane & 15;
  const int hi   = lane >> 4;
  const int mTile  = wave & 1;
  const int ntBase = wave >> 1;

  const _Float16* qsrc = qh + (size_t)bq * 32 * 128;
  const _Float16* dsrc = dh + (size_t)bd * 256 * 128;

  // ---- stage tiles into LDS -----------------------------------------------
#if CH_HAS_TDM
  if (wave == 0) {
    tdm_load_1d_padded((unsigned)LDS_DTILE_OFF, dsrc, 256u * 128u * 2u / 8u);
    tdm_load_1d_padded((unsigned)LDS_QTILE_OFF, qsrc, 32u * 128u * 2u / 8u);
    __builtin_amdgcn_s_wait_tensorcnt(0);
  }
#else
  for (int ch = tid; ch < 256 * 16; ch += 256) {        // d: 4096 x 16B
    const int row = ch >> 4, c = ch & 15;
    *(v8h*)(dtile + row * ROW_STRIDE + c * 8) = *(const v8h*)(dsrc + row * 128 + c * 8);
  }
  for (int ch = tid; ch < 32 * 16; ch += 256) {         // q: 512 x 16B
    const int row = ch >> 4, c = ch & 15;
    *(v8h*)(qtile + row * ROW_STRIDE + c * 8) = *(const v8h*)(qsrc + row * 128 + c * 8);
  }
#endif
  __syncthreads();

  // ---- A fragments from LDS (16x32 f16 / K-step); lane row M = l16 --------
  const _Float16* qrow = qtile + (mTile * 16 + l16) * ROW_STRIDE;
  const int asel = hi ? 8 : 0;
  v16h afrag[4];
#pragma unroll
  for (int ks = 0; ks < 4; ++ks) {
    v8h lo = *(const v8h*)(qrow + ks * 32 + asel);
    v8h hh = *(const v8h*)(qrow + ks * 32 + 16 + asel);
    afrag[ks] = __builtin_shufflevector(lo, hh,
        0, 1, 2, 3, 4, 5, 6, 7, 8, 9, 10, 11, 12, 13, 14, 15);
  }

  const int* qm = qmask + bq * 32;
  const int* dm = dmask + bd * 256;
  int qv[8];
#pragma unroll
  for (int r = 0; r < 8; ++r) qv[r] = qm[mTile * 16 + r + (hi ? 8 : 0)];

  const int bsel = hi ? 8 : 0;

#pragma unroll
  for (int i = 0; i < 4; ++i) {
    const int nt = ntBase + 4 * i;
    const _Float16* drow = dtile + (nt * 16 + l16) * ROW_STRIDE;
    // preload all 4 B fragments so the DS loads clause ahead of the WMMAs
    v16h bfrag[4];
#pragma unroll
    for (int ks = 0; ks < 4; ++ks) {
      v8h lo = *(const v8h*)(drow + ks * 32 + bsel);
      v8h hh = *(const v8h*)(drow + ks * 32 + 16 + bsel);
      bfrag[ks] = __builtin_shufflevector(lo, hh,
          0, 1, 2, 3, 4, 5, 6, 7, 8, 9, 10, 11, 12, 13, 14, 15);
    }
    v8f c = {0.f, 0.f, 0.f, 0.f, 0.f, 0.f, 0.f, 0.f};
#pragma unroll
    for (int ks = 0; ks < 4; ++ks)
      c = __builtin_amdgcn_wmma_f32_16x16x32_f16(false, afrag[ks], false, bfrag[ks],
                                                 (short)0, c, false, false);
    const int dv = dm[nt * 16 + l16];
#pragma unroll
    for (int r = 0; r < 8; ++r) {
      const int t = mTile * 16 + r + (hi ? 8 : 0);
      sims[t * 257 + nt * 16 + l16] = (qv[r] != 0 && dv != 0) ? c[r] : CH_NEG_INF;
    }
  }
  __syncthreads();

  // ---- d2q: one thread per doc-token column (max over 32 query rows) ------
  {
    float cmax = CH_NEG_INF;
#pragma unroll 4
    for (int t = 0; t < 32; ++t) cmax = fmaxf(cmax, sims[t * 257 + tid]);
    const bool v = cmax > CH_VALID_THRESH;
    redval[tid] = v ? cmax : 0.0f;
    redcnt[tid] = v ? 1.0f : 0.0f;
  }
  // ---- q2d partials: 8 threads per query row ------------------------------
  {
    const int row = tid & 31, part = tid >> 5;
    float rmax = CH_NEG_INF;
#pragma unroll 4
    for (int j = 0; j < 32; ++j)
      rmax = fmaxf(rmax, sims[row * 257 + part * 32 + j]);
    rowpart[part * 32 + row] = rmax;
  }
  __syncthreads();

  for (int off = 128; off > 0; off >>= 1) {
    if (tid < off) {
      redval[tid] += redval[tid + off];
      redcnt[tid] += redcnt[tid + off];
    }
    __syncthreads();
  }
  if (tid < 32) {
    float rm = rowpart[tid];
#pragma unroll
    for (int p = 1; p < 8; ++p) rm = fmaxf(rm, rowpart[p * 32 + tid]);
    const bool v = rm > CH_VALID_THRESH;
    rowval[tid] = v ? rm : 0.0f;
    rowcnt[tid] = v ? 1.0f : 0.0f;
  }
  __syncthreads();
  for (int off = 16; off > 0; off >>= 1) {
    if (tid < off) {
      rowval[tid] += rowval[tid + off];
      rowcnt[tid] += rowcnt[tid + off];
    }
    __syncthreads();
  }
  if (tid == 0) {
    const float q2d = rowval[0] / fmaxf(rowcnt[0], 1.0f);
    const float d2q = redval[0] / fmaxf(redcnt[0], 1.0f);
    out[bq * 64 + bd] = 0.5f * (q2d + d2q);
  }
}

// ---------------------------------------------------------------------------
extern "C" void kernel_launch(void* const* d_in, const int* in_sizes, int n_in,
                              void* d_out, int out_size, void* d_ws, size_t ws_size,
                              hipStream_t stream) {
  (void)in_sizes; (void)n_in; (void)out_size; (void)ws_size;
  const float* q  = (const float*)d_in[0];   // [64,32,128] f32
  const int*   qm = (const int*)d_in[1];     // [64,32]
  const float* d  = (const float*)d_in[2];   // [64,256,128] f32
  const int*   dm = (const int*)d_in[3];     // [64,256]
  float* out = (float*)d_out;                // [64,64]

  const int QROWS = 64 * 32;    // 2048
  const int DROWS = 64 * 256;   // 16384
  _Float16* qh = (_Float16*)d_ws;
  _Float16* dh = (_Float16*)((char*)d_ws + (size_t)QROWS * 128 * sizeof(_Float16));

  chamfer_prep<<<QROWS / 8, 256, 0, stream>>>(q, qm, qh, QROWS);
  chamfer_prep<<<DROWS / 8, 256, 0, stream>>>(d, dm, dh, DROWS);
  chamfer_pair<<<dim3(64, 64), 256, LDS_TOTAL, stream>>>(qh, dh, qm, dm, out);
}
